// BaseTriangleMultiplicativeUpdate_670014898470
// MI455X (gfx1250) — matrix operand order repaired and checked by reading, not compiled, not run on hardware
//
#include <hip/hip_runtime.h>

// ---------------------------------------------------------------- types
typedef __bf16 bf16_t;
typedef __attribute__((ext_vector_type(16))) __bf16 v16bf;
typedef __attribute__((ext_vector_type(8)))  __bf16 v8bf;
typedef __attribute__((ext_vector_type(4)))  __bf16 v4bf;
typedef __attribute__((ext_vector_type(8)))  float  v8f;

#define NSEQ 384
#define CDIM 128
#define MROWS (384*384)           // 147456

static __device__ __forceinline__ bf16_t f2bf(float f) {
  union { float f; unsigned u; } in; in.f = f;
  unsigned r = in.u + 0x7FFFu + ((in.u >> 16) & 1u);   // round-to-nearest-even
  union { unsigned short s; bf16_t b; } out;
  out.s = (unsigned short)(r >> 16);
  return out.b;
}
static __device__ __forceinline__ float bf2f(bf16_t b) {
  union { unsigned short s; bf16_t b; } in; in.b = b;
  union { unsigned u; float f; } out; out.u = ((unsigned)in.s) << 16;
  return out.f;
}
static __device__ __forceinline__ float sigmoidf_(float x) {
  return 1.0f / (1.0f + __expf(-x));
}
#define WMMA_BF16(A, B, C) \
  __builtin_amdgcn_wmma_f32_16x16x32_bf16(false, (A), false, (B), (short)0, (C), false, false)

// ------------------------------------------------------- 1) input layernorm
__global__ __launch_bounds__(256)
void ln_in_kernel(const float* __restrict__ z,
                  const float* __restrict__ w,
                  const float* __restrict__ bia,
                  bf16_t* __restrict__ x) {
  const long row  = (long)blockIdx.x * 8 + (threadIdx.x >> 5);
  const int  lane = threadIdx.x & 31;
  const float* zr = z + row * CDIM;
  float4 v = *(const float4*)(zr + lane * 4);
  float s = v.x + v.y + v.z + v.w;
  #pragma unroll
  for (int m = 16; m >= 1; m >>= 1) s += __shfl_xor(s, m, 32);
  const float mu = s * (1.0f / 128.0f);
  float d0 = v.x - mu, d1 = v.y - mu, d2 = v.z - mu, d3 = v.w - mu;
  float q = d0*d0 + d1*d1 + d2*d2 + d3*d3;
  #pragma unroll
  for (int m = 16; m >= 1; m >>= 1) q += __shfl_xor(q, m, 32);
  const float rstd = rsqrtf(q * (1.0f / 128.0f) + 1e-5f);
  const int c = lane * 4;
  v4bf pk;
  pk[0] = f2bf(d0 * rstd * w[c + 0] + bia[c + 0]);
  pk[1] = f2bf(d1 * rstd * w[c + 1] + bia[c + 1]);
  pk[2] = f2bf(d2 * rstd * w[c + 2] + bia[c + 2]);
  pk[3] = f2bf(d3 * rstd * w[c + 3] + bia[c + 3]);
  *(v4bf*)(x + row * CDIM + c) = pk;   // one 8B store
}

// ------------------------------------------- 2) fused dual GEMM + sigmoid gate
// D[c][row] = gated ? mask*sigmoid(x@wG+bG)*(x@wP+bP) : sigmoid(x@wG+bG)
// A-operand = W^T (staged transposed bf16 in LDS), B-operand = x rows.
// Software-pipelined (ping-pong) across row tiles.
__global__ __launch_bounds__(256)
void proj_kernel(const bf16_t* __restrict__ x,   // [MROWS][128]
                 const float* __restrict__ wG,   // [128k][128c]
                 const float* __restrict__ bG,
                 const float* __restrict__ wP,
                 const float* __restrict__ bP,
                 const float* __restrict__ mask, // [MROWS]
                 bf16_t* __restrict__ outp,      // [128c][MROWS]
                 int gated) {
  __shared__ bf16_t sWG[128 * 128];   // W^T: [c][k]
  __shared__ bf16_t sWP[128 * 128];
  const int tid = threadIdx.x;
  for (int idx = tid; idx < 128 * 128; idx += 256) {
    const int c = idx >> 7, k = idx & 127;
    sWG[idx] = f2bf(wG[k * 128 + c]);
    sWP[idx] = f2bf(wP[k * 128 + c]);
  }
  __syncthreads();

  const int wave = tid >> 5, lane = tid & 31;
  const int c0 = wave * 16;
  const int h = lane >> 4, l = lane & 15;

  v16bf aG[4], aP[4];
  #pragma unroll
  for (int kb = 0; kb < 4; ++kb) {
    const bf16_t* pG = sWG + (c0 + l) * 128 + kb * 32 + h * 16;
    const bf16_t* pP = sWP + (c0 + l) * 128 + kb * 32 + h * 16;
    union { v16bf v; v8bf hf[2]; } ug, up;
    ug.hf[0] = *(const v8bf*)pG;  ug.hf[1] = *(const v8bf*)(pG + 8);
    up.hf[0] = *(const v8bf*)pP;  up.hf[1] = *(const v8bf*)(pP + 8);
    aG[kb] = ug.v; aP[kb] = up.v;
  }
  float biasG[8], biasP[8];
  #pragma unroll
  for (int v = 0; v < 8; ++v) {
    biasG[v] = bG[c0 + h * 8 + v];
    biasP[v] = bP[c0 + h * 8 + v];
  }

  const long rowBase = (long)blockIdx.x * 128;
  const bf16_t* xbase = x + (rowBase + l) * 128 + h * 16;

  auto LDX = [&](int rt, v16bf bx[4]) {
    const bf16_t* xr = xbase + (size_t)rt * 16 * 128;
    bx[0] = *(const v16bf*)(xr);
    bx[1] = *(const v16bf*)(xr + 32);
    bx[2] = *(const v16bf*)(xr + 64);
    bx[3] = *(const v16bf*)(xr + 96);
  };
  auto COMPUTE = [&](int rt, const v16bf bx[4]) {
    v8f accG = {}; v8f accP = {};
    accG = WMMA_BF16(aG[0], bx[0], accG);  accP = WMMA_BF16(aP[0], bx[0], accP);
    accG = WMMA_BF16(aG[1], bx[1], accG);  accP = WMMA_BF16(aP[1], bx[1], accP);
    accG = WMMA_BF16(aG[2], bx[2], accG);  accP = WMMA_BF16(aP[2], bx[2], accP);
    accG = WMMA_BF16(aG[3], bx[3], accG);  accP = WMMA_BF16(aP[3], bx[3], accP);
    const long row = rowBase + rt * 16 + l;
    const float mrow = gated ? mask[row] : 1.0f;
    #pragma unroll
    for (int v = 0; v < 8; ++v) {
      const int c = c0 + h * 8 + v;
      const float gvv = sigmoidf_(accG[v] + biasG[v]);
      const float val = gated ? (mrow * gvv * (accP[v] + biasP[v])) : gvv;
      outp[(size_t)c * MROWS + row] = f2bf(val);
    }
  };

  v16bf bufA[4], bufB[4];
  LDX(0, bufA);
  #pragma unroll 1
  for (int rt = 0; rt < 8; rt += 2) {
    LDX(rt + 1, bufB);          // in flight during COMPUTE(rt)
    COMPUTE(rt, bufA);
    LDX((rt + 2) & 7, bufA);    // in flight during COMPUTE(rt+1); wrap = safe
    COMPUTE(rt + 1, bufB);
  }
}

// --------------------------------------------- 3) triangle einsum (NT GEMM/ch)
// t[c][i][j] = sum_k a[c][i][k]*b[c][j][k]; both frags are k-contiguous loads.
// Ping-pong double buffering across k-steps; 16i x 64j per wave.
__global__ __launch_bounds__(256)
void tri_kernel(const bf16_t* __restrict__ a,  // [128][384][384]
                const bf16_t* __restrict__ b,
                float* __restrict__ t) {       // [128][384][384]
  const int c  = blockIdx.z;
  const int i0 = blockIdx.x * 128 + (threadIdx.x >> 5) * 16;
  const int j0 = blockIdx.y * 64;
  const int lane = threadIdx.x & 31;
  const int h = lane >> 4, l = lane & 15;
  const bf16_t* pa = a + (size_t)c * MROWS + (size_t)(i0 + l) * NSEQ + h * 16;
  const bf16_t* pb = b + (size_t)c * MROWS + (size_t)(j0 + l) * NSEQ + h * 16;

  auto LDK = [&](int kb, v16bf& af, v16bf vb[4]) {
    union { v16bf v; v8bf hf[2]; } ua;
    ua.hf[0] = *(const v8bf*)(pa + kb * 32);
    ua.hf[1] = *(const v8bf*)(pa + kb * 32 + 8);
    af = ua.v;
    #pragma unroll
    for (int jt = 0; jt < 4; ++jt)
      vb[jt] = *(const v16bf*)(pb + (size_t)jt * 16 * NSEQ + kb * 32);
  };

  v8f acc[4] = {};
  v16bf afA, afB, vbA[4], vbB[4];
  LDK(0, afA, vbA);
  #pragma unroll 1
  for (int kb = 0; kb < 12; kb += 2) {
    LDK(kb + 1, afB, vbB);                      // hidden under compute A
    #pragma unroll
    for (int jt = 0; jt < 4; ++jt)
      acc[jt] = WMMA_BF16(afA, vbA[jt], acc[jt]);
    LDK((kb + 2 < 12) ? (kb + 2) : 0, afA, vbA); // hidden under compute B
    #pragma unroll
    for (int jt = 0; jt < 4; ++jt)
      acc[jt] = WMMA_BF16(afB, vbB[jt], acc[jt]);
  }
  float* T = t + (size_t)c * MROWS;
  #pragma unroll
  for (int jt = 0; jt < 4; ++jt) {
    #pragma unroll
    for (int v = 0; v < 8; ++v)
      T[(size_t)(i0 + v + h * 8) * NSEQ + j0 + jt * 16 + l] = acc[jt][v];
  }
}

// ------------------------------- 4) LN(t) @ w_o + b_o, gated, f32 output
__global__ __launch_bounds__(256)
void out_kernel(const float* __restrict__ t,   // [128][MROWS]
                const bf16_t* __restrict__ g,  // [128][MROWS]
                const float* __restrict__ lnw,
                const float* __restrict__ lnb,
                const float* __restrict__ wo,  // [128k][128c]
                const float* __restrict__ bo,
                float* __restrict__ out) {     // [MROWS][128]
  const int LSTR = 136;                        // bank-friendly, 16B-aligned rows
  __shared__ bf16_t sWT[128 * 128];            // w_o^T : [c][k]
  __shared__ bf16_t sLT[64 * 136];             // LN(t) rows : [row][k]
  __shared__ float  sRed[4 * 64 * 2];
  __shared__ float  sStat[64 * 2];
  const int tid = threadIdx.x;
  for (int idx = tid; idx < 128 * 128; idx += 256) {
    const int c = idx >> 7, k = idx & 127;
    sWT[idx] = f2bf(wo[k * 128 + c]);
  }
  const long row0 = (long)blockIdx.x * 64;
  {
    const int r = tid & 63, q = tid >> 6;      // 4 threads per row
    float s = 0.f, ss = 0.f;
    #pragma unroll 1
    for (int cc = 0; cc < 32; ++cc) {
      const float v = t[(size_t)(q * 32 + cc) * MROWS + row0 + r];
      s += v; ss += v * v;
    }
    sRed[(q * 64 + r) * 2 + 0] = s;
    sRed[(q * 64 + r) * 2 + 1] = ss;
    __syncthreads();
    if (q == 0) {
      float S = 0.f, SS = 0.f;
      #pragma unroll
      for (int qq = 0; qq < 4; ++qq) {
        S  += sRed[(qq * 64 + r) * 2 + 0];
        SS += sRed[(qq * 64 + r) * 2 + 1];
      }
      const float mu  = S * (1.0f / 128.0f);
      const float var = SS * (1.0f / 128.0f) - mu * mu;
      sStat[r * 2 + 0] = mu;
      sStat[r * 2 + 1] = rsqrtf(var + 1e-5f);
    }
    __syncthreads();
    const float mu = sStat[r * 2 + 0], rstd = sStat[r * 2 + 1];
    #pragma unroll 1
    for (int cc = 0; cc < 32; ++cc) {
      const int c = q * 32 + cc;
      const float v = t[(size_t)c * MROWS + row0 + r];
      sLT[r * LSTR + c] = f2bf((v - mu) * rstd * lnw[c] + lnb[c]);
    }
  }
  __syncthreads();

  const int wave = tid >> 5, lane = tid & 31;
  const int rt = wave & 3;           // row tile 0..3
  const int chalf = wave >> 2;       // c-tile half
  const int h = lane >> 4, l = lane & 15;

  v16bf afr[4];
  #pragma unroll
  for (int kb = 0; kb < 4; ++kb) {
    const bf16_t* p = sLT + (rt * 16 + l) * LSTR + kb * 32 + h * 16;
    union { v16bf v; v8bf hf[2]; } u;
    u.hf[0] = *(const v8bf*)p;
    u.hf[1] = *(const v8bf*)(p + 8);
    afr[kb] = u.v;
  }
  #pragma unroll 1
  for (int ci = 0; ci < 4; ++ci) {
    const int ct = chalf * 4 + ci;
    const bf16_t* pw = sWT + (ct * 16 + l) * 128 + h * 16;
    v16bf bw0 = *(const v16bf*)(pw);
    v16bf bw1 = *(const v16bf*)(pw + 32);
    v16bf bw2 = *(const v16bf*)(pw + 64);
    v16bf bw3 = *(const v16bf*)(pw + 96);
    v8f acc = {};
    acc = WMMA_BF16(afr[0], bw0, acc);
    acc = WMMA_BF16(afr[1], bw1, acc);
    acc = WMMA_BF16(afr[2], bw2, acc);
    acc = WMMA_BF16(afr[3], bw3, acc);
    #pragma unroll
    for (int v = 0; v < 8; ++v) {
      const long grow = row0 + rt * 16 + v + h * 8;
      const int  c    = ct * 16 + l;
      const float val  = acc[v] + bo[c];
      const float gate = bf2f(g[(size_t)c * MROWS + grow]);
      out[grow * 128 + c] = gate * val;
    }
  }
}

// ---------------------------------------------------------------- launcher
extern "C" void kernel_launch(void* const* d_in, const int* in_sizes, int n_in,
                              void* d_out, int out_size, void* d_ws, size_t ws_size,
                              hipStream_t stream) {
  const float* z     = (const float*)d_in[0];
  const float* mask  = (const float*)d_in[1];
  const float* lniw  = (const float*)d_in[2];
  const float* lnib  = (const float*)d_in[3];
  const float* w_a_p = (const float*)d_in[4];
  const float* b_a_p = (const float*)d_in[5];
  const float* w_a_g = (const float*)d_in[6];
  const float* b_a_g = (const float*)d_in[7];
  const float* w_b_p = (const float*)d_in[8];
  const float* b_b_p = (const float*)d_in[9];
  const float* w_b_g = (const float*)d_in[10];
  const float* b_b_g = (const float*)d_in[11];
  const float* w_g   = (const float*)d_in[12];
  const float* b_g   = (const float*)d_in[13];
  const float* lnow  = (const float*)d_in[14];
  const float* lnob  = (const float*)d_in[15];
  const float* w_o   = (const float*)d_in[16];
  const float* b_o   = (const float*)d_in[17];

  char* ws = (char*)d_ws;
  bf16_t* x  = (bf16_t*)ws;  ws += (size_t)MROWS * 128 * 2;
  bf16_t* av = (bf16_t*)ws;  ws += (size_t)MROWS * 128 * 2;
  bf16_t* bv = (bf16_t*)ws;  ws += (size_t)MROWS * 128 * 2;
  bf16_t* gv = (bf16_t*)ws;  ws += (size_t)MROWS * 128 * 2;
  float*  tv = (float*)ws;

  ln_in_kernel<<<MROWS / 8, 256, 0, stream>>>(z, lniw, lnib, x);
  proj_kernel<<<MROWS / 128, 256, 0, stream>>>(x, w_a_g, b_a_g, w_a_p, b_a_p, mask, av, 1);
  proj_kernel<<<MROWS / 128, 256, 0, stream>>>(x, w_b_g, b_b_g, w_b_p, b_b_p, mask, bv, 1);
  proj_kernel<<<MROWS / 128, 256, 0, stream>>>(x, w_g,   b_g,   w_g,   b_g,   mask, gv, 0);
  tri_kernel<<<dim3(3, 6, 128), 256, 0, stream>>>(av, bv, tv);
  out_kernel<<<MROWS / 64, 256, 0, stream>>>(tv, gv, lnow, lnob, w_o, b_o, (float*)d_out);
}
